// AttentionEncoder_14912126452058
// MI455X (gfx1250) — matrix operand order repaired and checked by reading, not compiled, not run on hardware
//
#include <hip/hip_runtime.h>
#include <hip/hip_bf16.h>

#define Bc  8
#define Sc  2048
#define Dc  128
#define Hc  2
#define DHc 64
#define NEGINF (-__builtin_huge_valf())

typedef __attribute__((ext_vector_type(16))) _Float16 v16h;
typedef __attribute__((ext_vector_type(8)))  float    v8f;
typedef unsigned int v4u __attribute__((ext_vector_type(4)));
typedef int          v8i __attribute__((ext_vector_type(8)));
typedef int          v4i __attribute__((ext_vector_type(4)));

#if __has_builtin(__builtin_amdgcn_tensor_load_to_lds) && \
    __has_builtin(__builtin_amdgcn_s_wait_tensorcnt)
#define USE_TDM 1
#else
#define USE_TDM 0
#endif

// ---- WMMA fragment index maps (CDNA5 ISA 7.12.2, wave32) ----
// A-matrix 16x32 f16: lane m=l&15 holds row m; k order used here (contiguous, matches B): lanes0-15 K=0..15, lanes16-31 K=16..31
__device__ __forceinline__ int kmapA(int g, int e) {
  int v = e >> 1, lo = e & 1;
  return 16 * g + 2 * v + lo;
}
// B-matrix 32x16 f16: lane n=l&15 holds col n; k order used here (contiguous, matches A): lanes0-15 K=0..15, lanes16-31 K=16..31
__device__ __forceinline__ int kmapB(int g, int e) {
  int v = e >> 1, lo = e & 1;
  return 16 * g + 2 * v + lo;
}

__device__ __forceinline__ v8f wmma32(v16h a, v16h b, v8f c) {
  return __builtin_amdgcn_wmma_f32_16x16x32_f16(false, a, false, b, (short)0, c,
                                                false, false);
}

// A-frag: A points at the first row of the 16-row tile, row stride ldk
__device__ __forceinline__ v16h load_a_frag(const _Float16* A, int ldk,
                                            int k_base, int lane) {
  int m = lane & 15, g = lane >> 4;
  const _Float16* p = A + (size_t)m * ldk + k_base;
  v16h f;
#pragma unroll
  for (int e = 0; e < 16; ++e) f[e] = p[kmapA(g, e)];
  return f;
}
// B[k][n] = W[(n_base+n)*ldk + k_base + k]  (k contiguous per lane -> b128)
__device__ __forceinline__ v16h load_b_frag_row(const _Float16* W, int n_base,
                                                int ldk, int k_base, int lane) {
  int n = lane & 15, g = lane >> 4;
  const _Float16* p = W + (size_t)(n_base + n) * ldk + k_base;
  v16h f;
#pragma unroll
  for (int e = 0; e < 16; ++e) f[e] = p[kmapB(g, e)];
  return f;
}

#if USE_TDM
// Issue a 2-D TDM tile load: rows x row_elems f16, row stride stride_elems,
// global -> LDS byte address lds_byte. Descriptor per CDNA5 ISA ch.8 (D#).
__device__ __forceinline__ void tdm_load_tile_f16(unsigned int lds_byte,
                                                  const _Float16* gsrc,
                                                  int row_elems, int rows,
                                                  int stride_elems) {
  unsigned long long ga = (unsigned long long)(uintptr_t)gsrc;
  v4u g0;
  g0[0] = 1u;                                   // count=1, user mode, no gather
  g0[1] = lds_byte;                             // lds_addr
  g0[2] = (unsigned int)ga;                     // global_addr[31:0]
  g0[3] = (unsigned int)((ga >> 32) & 0x01ffffffu) | 0x80000000u;  // [56:32] | type=2
  v8i g1;
  g1[0] = (int)(1u << 16);                      // wg_mask=0, data_size=1 (2 bytes)
  g1[1] = (int)(((unsigned)row_elems & 0xffffu) << 16);            // tensor_dim0 lo @bit48
  g1[2] = (int)((((unsigned)row_elems >> 16) & 0xffffu) |
                (((unsigned)rows & 0xffffu) << 16));               // dim0 hi | tensor_dim1 lo
  g1[3] = (int)((((unsigned)rows >> 16) & 0xffffu) |
                (((unsigned)row_elems & 0xffffu) << 16));          // dim1 hi | tile_dim0
  g1[4] = (int)((unsigned)rows & 0xffffu);      // tile_dim1 | tile_dim2=0
  g1[5] = stride_elems;                         // tensor_dim0_stride[31:0]
  g1[6] = 0;                                    // stride hi | dim1_stride lo
  g1[7] = 0;
  v4i gz = {0, 0, 0, 0};
#if defined(__clang_major__) && __clang_major__ >= 23
  v8i z8 = {0, 0, 0, 0, 0, 0, 0, 0};
  __builtin_amdgcn_tensor_load_to_lds(g0, g1, gz, gz, z8, 0);
#else
  __builtin_amdgcn_tensor_load_to_lds(g0, g1, gz, gz, 0);
#endif
}
#endif

// ---------------- fp32 -> f16 convert ----------------
__global__ void f32_to_f16(const float* __restrict__ s, _Float16* __restrict__ d,
                           int n) {
  int i = blockIdx.x * 256 + threadIdx.x;
  if (i < n) d[i] = (_Float16)s[i];
}

// ---------------- fused dual layernorm: ln(x+ate) and ln(x) ----------------
__global__ __launch_bounds__(256) void ln_kernel(
    const float* __restrict__ x, const float* __restrict__ ate,
    const float* __restrict__ g, const float* __restrict__ b,
    _Float16* __restrict__ qk_h, _Float16* __restrict__ v_h) {
  int wave = threadIdx.x >> 5, lane = threadIdx.x & 31;
  int row = blockIdx.x * 8 + wave;  // B*S rows
  const float* xp = x + (size_t)row * Dc;
  const float* ap = ate + (size_t)row * Dc;
  float xv[4], qv[4];
  float sx = 0.f, sxx = 0.f, sq = 0.f, sqq = 0.f;
#pragma unroll
  for (int i = 0; i < 4; ++i) {
    float xi = xp[lane * 4 + i];
    float qi = xi + ap[lane * 4 + i];
    xv[i] = xi; qv[i] = qi;
    sx += xi; sxx += xi * xi; sq += qi; sqq += qi * qi;
  }
#pragma unroll
  for (int o = 16; o > 0; o >>= 1) {
    sx += __shfl_xor(sx, o, 32);  sxx += __shfl_xor(sxx, o, 32);
    sq += __shfl_xor(sq, o, 32);  sqq += __shfl_xor(sqq, o, 32);
  }
  float mx = sx * (1.f / Dc), vx = sxx * (1.f / Dc) - mx * mx;
  float mq = sq * (1.f / Dc), vq = sqq * (1.f / Dc) - mq * mq;
  float rx = rsqrtf(vx + 1e-5f), rq = rsqrtf(vq + 1e-5f);
#pragma unroll
  for (int i = 0; i < 4; ++i) {
    int c = lane * 4 + i;
    float gg = g[c], bb = b[c];
    v_h[(size_t)row * Dc + c]  = (_Float16)((xv[i] - mx) * rx * gg + bb);
    qk_h[(size_t)row * Dc + c] = (_Float16)((qv[i] - mq) * rq * gg + bb);
  }
}

// ---------------- generic WMMA GEMM + epilogues ----------------
// out[m][n] = sum_k A[m][k] * W[n][k]   (W row-major [N,K])
// MODE 0: QKV scatter -> o0=q [B,H,S,DH] (pre-scaled 1/8), o1=k [B,H,S,DH],
//         o2 = v TRANSPOSED [B,H,DH,S] (for contiguous PV fragments)
// MODE 1: x2 = resid + gemm + bias -> outf (f32) and o0 (f16)
// MODE 2: o0 = (f16)(2*relu(gemm + bias))
// MODE 3: outf = resid + gemm + bias   (final output)
template <int MODE, int N, int K>
__global__ __launch_bounds__(256) void gemm_ep(
    const _Float16* __restrict__ A0, const _Float16* __restrict__ A1,
    const _Float16* __restrict__ W, const float* __restrict__ bias,
    const float* __restrict__ resid, float* __restrict__ outf,
    _Float16* __restrict__ o0, _Float16* __restrict__ o1,
    _Float16* __restrict__ o2) {
  constexpr int NK = K / 32;
  int wave = threadIdx.x >> 5, lane = threadIdx.x & 31;
  int mt = blockIdx.x * 8 + wave;  // 1024 tiles of 16 rows
  int row0 = mt * 16;
  int g = lane >> 4, nl = lane & 15;

  v16h af[NK];
#pragma unroll
  for (int kt = 0; kt < NK; ++kt)
    af[kt] = load_a_frag(A0 + (size_t)row0 * K, K, kt * 32, lane);
  v16h af2[NK];
  if constexpr (MODE == 0) {
#pragma unroll
    for (int kt = 0; kt < NK; ++kt)
      af2[kt] = load_a_frag(A1 + (size_t)row0 * K, K, kt * 32, lane);
  }

  for (int nt = 0; nt < N / 16; ++nt) {
    v8f c = {};
#pragma unroll
    for (int kt = 0; kt < NK; ++kt) {
      v16h bf = load_b_frag_row(W, nt * 16, K, kt * 32, lane);
      v16h av;
      if constexpr (MODE == 0) av = (nt >= 16) ? af2[kt] : af[kt];
      else av = af[kt];
      c = wmma32(av, bf, c);
    }
#pragma unroll
    for (int r = 0; r < 8; ++r) {
      int m = r + 8 * g;
      int row = row0 + m;
      int col = nt * 16 + nl;
      float v = c[r] + bias[col];
      if constexpr (MODE == 0) {
        int b = row >> 11, s = row & (Sc - 1);
        if (col < Dc) {
          int h = col >> 6, dh = col & 63;
          o0[(((size_t)b * Hc + h) * Sc + s) * DHc + dh] = (_Float16)(v * 0.125f);
        } else if (col < 2 * Dc) {
          int e = col - Dc; int h = e >> 6, dh = e & 63;
          o1[(((size_t)b * Hc + h) * Sc + s) * DHc + dh] = (_Float16)v;
        } else {
          int e = col - 2 * Dc; int h = e >> 6, dh = e & 63;
          o2[(((size_t)b * Hc + h) * DHc + dh) * Sc + s] = (_Float16)v;  // V^T
        }
      } else if constexpr (MODE == 1) {
        float x2v = resid[(size_t)row * Dc + col] + v;
        outf[(size_t)row * Dc + col] = x2v;
        o0[(size_t)row * Dc + col] = (_Float16)x2v;
      } else if constexpr (MODE == 2) {
        float hv = 2.f * fmaxf(v, 0.f);
        o0[(size_t)row * N + col] = (_Float16)hv;
      } else {
        outf[(size_t)row * Dc + col] = resid[(size_t)row * Dc + col] + v;
      }
    }
  }
}

// ---------------- flash attention ----------------
// One wave per (b,h,16 q-rows); the 8 waves of a block share (b,h) and the
// key sequence, so per 32-key chunk K [32 key][64 dh] and V^T [64 dh][32 key]
// tiles (4KB each) are staged in LDS once per block via the Tensor Data
// Mover, double-buffered (prefetch kb+1 while computing kb).  Both K and V
// fragments then read k-contiguous per lane -> ds_load_b128.
// Dynamic-LDS layout (bytes):
//   [0,8K)    K chunks   [2][32*64] f16
//   [8K,16K)  V^T chunks [2][64*32] f16
//   [16K,24K) per-wave P repack [8][16*32] f16
__global__ __launch_bounds__(256) void attn_kernel(
    const _Float16* __restrict__ qg, const _Float16* __restrict__ kg,
    const _Float16* __restrict__ vg, const int* __restrict__ nodeT,
    const int* __restrict__ actions, _Float16* __restrict__ ctx) {
  extern __shared__ char smem[];
  _Float16* kbuf = (_Float16*)smem;             // [2][2048]
  _Float16* vbuf = (_Float16*)(smem + 8192);    // [2][2048] (V^T tiles)
  _Float16* pbuf = (_Float16*)(smem + 16384);   // [8][512]

  int wave = threadIdx.x >> 5, lane = threadIdx.x & 31;
  int wid = blockIdx.x * 8 + wave;     // 0..2047
  int qt = wid & (Sc / 16 - 1);        // 128 q-tiles
  int bh = wid >> 7;                   // b*H + h  (uniform across the block)
  int b = bh >> 1, h = bh & 1;
  int g = lane >> 4, nl = lane & 15;

  const _Float16* Qb = qg + (size_t)bh * Sc * DHc;
  const _Float16* Kb = kg + (size_t)bh * Sc * DHc;
  const _Float16* Vb = vg + (size_t)bh * DHc * Sc;  // V^T: [DH][S]

  v16h qf0 = load_a_frag(Qb + (size_t)qt * 16 * DHc, DHc, 0, lane);
  v16h qf1 = load_a_frag(Qb + (size_t)qt * 16 * DHc, DHc, 32, lane);

  // per-lane metadata for its 8 rows (rows m = r + 8*g)
  int qmac[8], qjob[8], qmaskon[8];
#pragma unroll
  for (int r = 0; r < 8; ++r) {
    int qr = qt * 16 + r + 8 * g;
    qmac[r] = nodeT[qr * 3 + 2];
    qjob[r] = nodeT[qr * 3 + 0];
    qmaskon[r] = (actions[(size_t)b * Sc + qr] == 0);
  }

  float mi[8], li[8];
  v8f acc[4];
#pragma unroll
  for (int r = 0; r < 8; ++r) { mi[r] = NEGINF; li[r] = 0.f; }
#pragma unroll
  for (int t = 0; t < 4; ++t) acc[t] = (v8f){};

#if USE_TDM
  unsigned int lds0 = __builtin_amdgcn_groupstaticsize();  // dynamic-LDS base
  if (wave == 0) {  // prefetch chunk 0 (EXEC is ignored by TDM; one wave issues)
    tdm_load_tile_f16(lds0 + 0u, Kb, DHc, 32, DHc);         // K: 32 rows x 64
    tdm_load_tile_f16(lds0 + 8192u, Vb, 32, DHc, Sc);       // V^T: 64 rows x 32
  }
#endif

  for (int kb = 0; kb < Sc / 32; ++kb) {
    int key0 = kb * 32;
    int cur = kb & 1;
#if USE_TDM
    if (wave == 0) {
      if (kb + 1 < Sc / 32) {  // prefetch next chunk into the other buffer
        const _Float16* Kn = Kb + (size_t)(kb + 1) * 32 * DHc;
        const _Float16* Vn = Vb + (size_t)(kb + 1) * 32;
        tdm_load_tile_f16(lds0 + (unsigned)((cur ^ 1) * 4096), Kn, DHc, 32, DHc);
        tdm_load_tile_f16(lds0 + 8192u + (unsigned)((cur ^ 1) * 4096), Vn, 32, DHc, Sc);
        __builtin_amdgcn_s_wait_tensorcnt(2);  // in-order: current chunk done
      } else {
        __builtin_amdgcn_s_wait_tensorcnt(0);
      }
    }
    __syncthreads();  // staged chunk visible to all 8 waves
    const _Float16* kc = kbuf + cur * 2048;
    const _Float16* vc = vbuf + cur * 2048;
#else
    __syncthreads();  // previous iteration finished reading the buffers
    {
      // K: contiguous 4KB; V^T: 64 rows of 32 halves, global row stride S
      ((uint4*)kbuf)[threadIdx.x] =
          ((const uint4*)(Kb + (size_t)key0 * DHc))[threadIdx.x];
      ((uint4*)vbuf)[threadIdx.x] =
          *(const uint4*)(Vb + (size_t)(threadIdx.x >> 2) * Sc + key0 +
                          (threadIdx.x & 3) * 8);
    }
    __syncthreads();
    const _Float16* kc = kbuf;
    const _Float16* vc = vbuf;
#endif

    // scores: two 16x16 tiles from LDS, K=64 contraction in two steps
    v8f s0 = {}, s1 = {};
    s0 = wmma32(qf0, load_b_frag_row(kc, 0, DHc, 0, lane), s0);
    s0 = wmma32(qf1, load_b_frag_row(kc, 0, DHc, 32, lane), s0);
    s1 = wmma32(qf0, load_b_frag_row(kc, 16, DHc, 0, lane), s1);
    s1 = wmma32(qf1, load_b_frag_row(kc, 16, DHc, 32, lane), s1);

    int k0 = key0 + nl, k1 = key0 + 16 + nl;
    int kmac0 = nodeT[k0 * 3 + 2], kjob0 = nodeT[k0 * 3 + 0];
    int kmac1 = nodeT[k1 * 3 + 2], kjob1 = nodeT[k1 * 3 + 0];

#pragma unroll
    for (int r = 0; r < 8; ++r) {
      if (qmaskon[r] && (qmac[r] != kmac0) && (qjob[r] != kjob0)) s0[r] = NEGINF;
      if (qmaskon[r] && (qmac[r] != kmac1) && (qjob[r] != kjob1)) s1[r] = NEGINF;
      // row max across the 16 lanes of this half (offsets 1,2,4,8 stay in half)
      float mx = fmaxf(s0[r], s1[r]);
#pragma unroll
      for (int o = 8; o > 0; o >>= 1) mx = fmaxf(mx, __shfl_xor(mx, o, 32));
      float mn = fmaxf(mi[r], mx);
      float msh = (mn == NEGINF) ? 0.f : mn;  // fully-masked chunk guard
      float al = __expf(mi[r] - msh);
      float p0 = __expf(s0[r] - msh);
      float p1 = __expf(s1[r] - msh);
      float rs = p0 + p1;
#pragma unroll
      for (int o = 8; o > 0; o >>= 1) rs += __shfl_xor(rs, o, 32);
      li[r] = li[r] * al + rs;
      mi[r] = mn;
#pragma unroll
      for (int t = 0; t < 4; ++t) acc[t][r] *= al;
      int m = r + 8 * g;
      pbuf[wave * 512 + m * 32 + nl] = (_Float16)p0;
      pbuf[wave * 512 + m * 32 + 16 + nl] = (_Float16)p1;
    }
    // repack P(16x32) into A-fragment layout from LDS (same-wave DS is in-order)
    v16h pf;
#pragma unroll
    for (int e = 0; e < 16; ++e) pf[e] = pbuf[wave * 512 + nl * 32 + kmapA(g, e)];
    // PV: B[k][n] = vT[(t*16+n)*32 + k] -> k-contiguous per lane
#pragma unroll
    for (int t = 0; t < 4; ++t)
      acc[t] = wmma32(pf, load_b_frag_row(vc, t * 16, 32, 0, lane), acc[t]);

#if USE_TDM
    __syncthreads();  // reads done before next iteration's prefetch overwrites
#endif
  }

#pragma unroll
  for (int r = 0; r < 8; ++r) {
    float inv = 1.f / li[r];
    int m = r + 8 * g;
    int s = qt * 16 + m;
#pragma unroll
    for (int t = 0; t < 4; ++t) {
      int d = h * DHc + t * 16 + nl;
      ctx[((size_t)b * Sc + s) * Dc + d] = (_Float16)(acc[t][r] * inv);
    }
  }
}

extern "C" void kernel_launch(void* const* d_in, const int* in_sizes, int n_in,
                              void* d_out, int out_size, void* d_ws,
                              size_t ws_size, hipStream_t stream) {
  (void)in_sizes; (void)n_in; (void)out_size; (void)ws_size;
  const float* x         = (const float*)d_in[0];
  const float* ate       = (const float*)d_in[1];
  const int*   actions   = (const int*)d_in[2];
  const int*   nodeT     = (const int*)d_in[3];
  const float* in_proj_w = (const float*)d_in[4];
  const float* in_proj_b = (const float*)d_in[5];
  const float* out_w     = (const float*)d_in[6];
  const float* out_b     = (const float*)d_in[7];
  const float* ln_g      = (const float*)d_in[8];
  const float* ln_b      = (const float*)d_in[9];
  const float* l1_w      = (const float*)d_in[10];
  const float* l1_b      = (const float*)d_in[11];
  const float* l2_w      = (const float*)d_in[12];
  const float* l2_b      = (const float*)d_in[13];
  float* out = (float*)d_out;

  char* ws = (char*)d_ws;
  size_t off = 0;
  auto alloc = [&](size_t bytes) {
    size_t r = off;
    off += (bytes + 255) & ~(size_t)255;
    return r;
  };
  const size_t MS = (size_t)Bc * Sc;  // 16384 rows
  _Float16* qk_h   = (_Float16*)(ws + alloc(MS * Dc * 2));
  _Float16* vin_h  = (_Float16*)(ws + alloc(MS * Dc * 2));
  _Float16* wqkv_h = (_Float16*)(ws + alloc((size_t)3 * Dc * Dc * 2));
  _Float16* wout_h = (_Float16*)(ws + alloc((size_t)Dc * Dc * 2));
  _Float16* wl1_h  = (_Float16*)(ws + alloc((size_t)2 * Dc * Dc * 2));
  _Float16* wl2_h  = (_Float16*)(ws + alloc((size_t)Dc * 2 * Dc * 2));
  _Float16* q_h    = (_Float16*)(ws + alloc(MS * Dc * 2));
  _Float16* k_h    = (_Float16*)(ws + alloc(MS * Dc * 2));
  _Float16* v_h    = (_Float16*)(ws + alloc(MS * Dc * 2));  // V^T [B,H,DH,S]
  _Float16* ctx_h  = (_Float16*)(ws + alloc(MS * Dc * 2));
  float*    x2f    = (float*)(ws + alloc(MS * Dc * 4));
  _Float16* x2_h   = (_Float16*)(ws + alloc(MS * Dc * 2));
  _Float16* hh     = (_Float16*)(ws + alloc(MS * 2 * Dc * 2));

  // weight converts
  f32_to_f16<<<(3 * Dc * Dc + 255) / 256, 256, 0, stream>>>(in_proj_w, wqkv_h, 3 * Dc * Dc);
  f32_to_f16<<<(Dc * Dc + 255) / 256, 256, 0, stream>>>(out_w, wout_h, Dc * Dc);
  f32_to_f16<<<(2 * Dc * Dc + 255) / 256, 256, 0, stream>>>(l1_w, wl1_h, 2 * Dc * Dc);
  f32_to_f16<<<(2 * Dc * Dc + 255) / 256, 256, 0, stream>>>(l2_w, wl2_h, 2 * Dc * Dc);

  // dual layernorm
  ln_kernel<<<MS / 8, 256, 0, stream>>>(x, ate, ln_g, ln_b, qk_h, vin_h);

  // QKV projection (N=384, K=128); q pre-scaled by 1/sqrt(64); v transposed
  gemm_ep<0, 384, 128><<<128, 256, 0, stream>>>(qk_h, vin_h, wqkv_h, in_proj_b,
                                                nullptr, nullptr, q_h, k_h, v_h);
  // masked flash attention with TDM-staged K/V^T (24 KB dynamic LDS)
  attn_kernel<<<256, 256, 24576, stream>>>(q_h, k_h, v_h, nodeT, actions, ctx_h);

  // out projection + residual -> x2
  gemm_ep<1, 128, 128><<<128, 256, 0, stream>>>(ctx_h, nullptr, wout_h, out_b,
                                                x, x2f, x2_h, nullptr, nullptr);
  // MLP1: 2*relu
  gemm_ep<2, 256, 128><<<128, 256, 0, stream>>>(x2_h, nullptr, wl1_h, l1_b,
                                                nullptr, nullptr, hh, nullptr, nullptr);
  // MLP2 + residual -> final output
  gemm_ep<3, 128, 256><<<128, 256, 0, stream>>>(hh, nullptr, wl2_h, l2_b,
                                                x2f, out, nullptr, nullptr, nullptr);
}
